// ResnetBlockReluConvLattice_28664611733903
// MI455X (gfx1250) — matrix operand order repaired and checked by reading, not compiled, not run on hardware
//
#include <hip/hip_runtime.h>
#include <hip/hip_bf16.h>

// ---------------------------------------------------------------------------
// ResNet block of two permutohedral-lattice convolutions, CDNA5 (gfx1250).
// N=300000 vertices, K=9 taps, F=64 filters.  BF16 WMMA (16x16x32) with f32
// accumulate.  Weights staged in LDS (ds_load path), activation gathers from
// global with up-front prefetch (global_prefetch_b8).
// ---------------------------------------------------------------------------

typedef __attribute__((ext_vector_type(16))) __bf16 v16bf;
typedef __attribute__((ext_vector_type(8)))  float  v8f;

#define NVERT  300000
#define KTAP   9
#define FEAT   64
#define RTOT   (KTAP * FEAT)       // 576 reduction length
#define NTILES (NVERT / 16)        // 18750, exact
#define LDSTR  (RTOT + 8)          // 584 bf16: padded LDS row stride (bank-spread)

// round-to-nearest-even float -> bf16 bits
static __device__ __forceinline__ unsigned short f2bf(float f) {
  unsigned u = __builtin_bit_cast(unsigned, f);
  u += 0x7FFFu + ((u >> 16) & 1u);
  return (unsigned short)(u >> 16);
}

// ---------------------------------------------------------------------------
// Kernel 0a: xbf = bf16(relu(lv)), 4 elements / thread (19.2M total, /4 exact)
// ---------------------------------------------------------------------------
__global__ void relu_cvt_kernel(const float* __restrict__ x,
                                unsigned short* __restrict__ y, int n4) {
  int i = blockIdx.x * blockDim.x + threadIdx.x;
  if (i >= n4) return;
  const float4 v = *(const float4*)(x + (size_t)i * 4);
  unsigned a = (unsigned)f2bf(fmaxf(v.x, 0.f)) | ((unsigned)f2bf(fmaxf(v.y, 0.f)) << 16);
  unsigned b = (unsigned)f2bf(fmaxf(v.z, 0.f)) | ((unsigned)f2bf(fmaxf(v.w, 0.f)) << 16);
  uint2 o; o.x = a; o.y = b;
  *(uint2*)(y + (size_t)i * 4) = o;
}

// ---------------------------------------------------------------------------
// Kernel 0b: wT[o*576 + r] = bf16( w[r*64 + o] )   (r = tap*64 + f_in)
// ---------------------------------------------------------------------------
__global__ void cvt_wT_kernel(const float* __restrict__ w,
                              unsigned short* __restrict__ wT) {
  int i = blockIdx.x * blockDim.x + threadIdx.x;   // i = o*576 + r
  if (i >= RTOT * FEAT) return;
  int o = i / RTOT;
  int r = i - o * RTOT;
  wT[i] = f2bf(w[(size_t)r * FEAT + o]);
}

// ---------------------------------------------------------------------------
// Fragment loaders (wave32, contiguous k order shared by A and B)
//   A 16x32 bf16: lane L holds row M=L%16; 16 contiguous K at
//   (L<16 ? 0 : 16).  Two global 16B loads.
//   B 32x16 bf16: lane L holds col N=L%16; 16 contiguous K at
//   (L<16 ? 0 : 16).  Two LDS 16B loads (contiguous 32B).
// ---------------------------------------------------------------------------
static __device__ __forceinline__ v16bf load_a_frag(const unsigned short* __restrict__ row,
                                                    int koff, int lane) {
  const int sel = (lane & 16) ? 16 : 0;
  union { v16bf bf; uint4 q[2]; } u;
  u.q[0] = *(const uint4*)(row + koff + sel);
  u.q[1] = *(const uint4*)(row + koff + sel + 8);
  return u.bf;
}

static __device__ __forceinline__ v16bf load_b_frag_lds(const unsigned short* smw,
                                                        int n, int r0, int lane) {
  const int sel = (lane & 16) ? 16 : 0;
  const unsigned short* p = smw + (unsigned)n * LDSTR + r0 + sel;
  union { v16bf bf; uint4 q[2]; } u;
  u.q[0] = ((const uint4*)p)[0];
  u.q[1] = ((const uint4*)p)[1];
  return u.bf;
}

// ---------------------------------------------------------------------------
// Lattice GEMM: one wave32 per 16-vertex tile, full 16x64 output in 4 C tiles.
// Weights staged in LDS once per block (74.75 KB, padded rows).
// LAYER2=0: out_bf = bf16(relu(acc + bias))      (feeds layer 2)
// LAYER2=1: out_f  = acc + bias + lv             (residual, f32)
// ---------------------------------------------------------------------------
template <int LAYER2>
__global__ void lattice_gemm_kernel(const unsigned short* __restrict__ xbf,
                                    const long long* __restrict__ nbrs,
                                    const unsigned short* __restrict__ wT,
                                    const float* __restrict__ bias,
                                    unsigned short* __restrict__ out_bf,
                                    const float* __restrict__ lv,
                                    float* __restrict__ out_f) {
  __shared__ unsigned short smw[FEAT * LDSTR];     // 64 * 584 * 2B = 74752 B

  // --- cooperative weight staging: 36864 bf16 in 8-element chunks ----------
  // 8 | 576, so a chunk never crosses a row; re-index into padded rows.
  for (int idx = threadIdx.x; idx < (RTOT * FEAT) / 8; idx += blockDim.x) {
    const int e0 = idx * 8;
    const int n  = e0 / RTOT;
    const int r  = e0 - n * RTOT;
    *(uint4*)(smw + (unsigned)n * LDSTR + r) = *(const uint4*)(wT + e0);
  }
  __syncthreads();

  const int tile = blockIdx.x * (blockDim.x >> 5) + (threadIdx.x >> 5);
  const int lane = threadIdx.x & 31;
  const int m16  = lane & 15;

  if (tile < NTILES) {                             // wave-uniform guard
    const long long mrow = (long long)tile * 16 + m16;

    // hoist the 9 neighbor indices (contiguous 72B per lane) + row pointers,
    // then prefetch every gathered row so lines are in flight before the MACs.
    const unsigned short* rowp[KTAP];
#pragma unroll
    for (int tap = 0; tap < KTAP; ++tap) {
      const long long nb = nbrs[mrow * KTAP + tap];
      rowp[tap] = xbf + nb * FEAT;
    }
#pragma unroll
    for (int tap = 0; tap < KTAP; ++tap) {
      __builtin_prefetch(rowp[tap], 0, 3);                 // global_prefetch_b8
      __builtin_prefetch((const char*)rowp[tap] + 64, 0, 3);
    }

    v8f acc0 = {}, acc1 = {}, acc2 = {}, acc3 = {};

    for (int tap = 0; tap < KTAP; ++tap) {
#pragma unroll
      for (int h = 0; h < 2; ++h) {
        const int koff = h * 32;
        const v16bf a  = load_a_frag(rowp[tap], koff, lane);
        const int  r0  = tap * FEAT + koff;
        const v16bf b0 = load_b_frag_lds(smw,  0 + m16, r0, lane);
        const v16bf b1 = load_b_frag_lds(smw, 16 + m16, r0, lane);
        const v16bf b2 = load_b_frag_lds(smw, 32 + m16, r0, lane);
        const v16bf b3 = load_b_frag_lds(smw, 48 + m16, r0, lane);
        acc0 = __builtin_amdgcn_wmma_f32_16x16x32_bf16(false, a, false, b0, (short)0, acc0, false, false);
        acc1 = __builtin_amdgcn_wmma_f32_16x16x32_bf16(false, a, false, b1, (short)0, acc1, false, false);
        acc2 = __builtin_amdgcn_wmma_f32_16x16x32_bf16(false, a, false, b2, (short)0, acc2, false, false);
        acc3 = __builtin_amdgcn_wmma_f32_16x16x32_bf16(false, a, false, b3, (short)0, acc3, false, false);
      }
    }

    // Epilogue.  C layout: VGPR i, lanes 0-15 -> M=i, lanes 16-31 -> M=8+i;
    // N = lane%16.
    const int mlo = (lane & 16) ? 8 : 0;
    v8f accs[4] = {acc0, acc1, acc2, acc3};
#pragma unroll
    for (int j = 0; j < 4; ++j) {
      const int n   = j * 16 + m16;
      const float bv = bias[n];
#pragma unroll
      for (int i = 0; i < 8; ++i) {
        const long long m = (long long)tile * 16 + mlo + i;
        const float v = accs[j][i] + bv;
        if (LAYER2) {
          out_f[m * FEAT + n] = v + lv[m * FEAT + n];
        } else {
          out_bf[m * FEAT + n] = f2bf(fmaxf(v, 0.f));
        }
      }
    }
  }
}

// ---------------------------------------------------------------------------
// Launch
// ---------------------------------------------------------------------------
extern "C" void kernel_launch(void* const* d_in, const int* in_sizes, int n_in,
                              void* d_out, int out_size, void* d_ws, size_t ws_size,
                              hipStream_t stream) {
  const float*     lv   = (const float*)d_in[0];      // [N,64] f32
  const long long* nbrs = (const long long*)d_in[1];  // [N,9]  i64
  const float*     w1   = (const float*)d_in[2];      // [9,64,64] f32
  const float*     b1   = (const float*)d_in[3];      // [64] f32
  const float*     w2   = (const float*)d_in[4];
  const float*     b2   = (const float*)d_in[5];
  float*           out  = (float*)d_out;              // [N,64] f32

  // Workspace layout (bytes, 256-aligned)
  char* ws = (char*)d_ws;
  const size_t act_bytes = (size_t)NVERT * FEAT * sizeof(unsigned short); // 38.4 MB
  const size_t wt_bytes  = (size_t)RTOT * FEAT * sizeof(unsigned short);  // 72 KB
  unsigned short* xbf1 = (unsigned short*)(ws);
  unsigned short* xbf2 = (unsigned short*)(ws + act_bytes);
  unsigned short* wT1  = (unsigned short*)(ws + 2 * act_bytes);
  unsigned short* wT2  = (unsigned short*)(ws + 2 * act_bytes + ((wt_bytes + 255) & ~255ull));
  (void)ws_size; (void)in_sizes; (void)n_in; (void)out_size;

  // 0a: bf16(relu(lv))
  {
    const int n4 = NVERT * FEAT / 4;                  // exact
    relu_cvt_kernel<<<(n4 + 255) / 256, 256, 0, stream>>>(lv, xbf1, n4);
  }
  // 0b: transpose+convert weights
  {
    const int ne = RTOT * FEAT;
    cvt_wT_kernel<<<(ne + 255) / 256, 256, 0, stream>>>(w1, wT1);
    cvt_wT_kernel<<<(ne + 255) / 256, 256, 0, stream>>>(w2, wT2);
  }
  // Layer 1: xbf2 = bf16(relu(conv(xbf1, w1) + b1))
  const int waves_per_block = 8;                      // 256 threads
  const int blocks = (NTILES + waves_per_block - 1) / waves_per_block;
  lattice_gemm_kernel<0><<<blocks, 256, 0, stream>>>(xbf1, nbrs, wT1, b1,
                                                     xbf2, nullptr, nullptr);
  // Layer 2: out = conv(xbf2, w2) + b2 + lv
  lattice_gemm_kernel<1><<<blocks, 256, 0, stream>>>(xbf2, nbrs, wT2, b2,
                                                     nullptr, lv, out);
}
